// CorrBlock_69595650064956
// MI455X (gfx1250) — matrix operand order repaired and checked by reading, not compiled, not run on hardware
//
#include <hip/hip_runtime.h>

// corr[n,m] = <fmap1[n], fmap2[m]> / sqrt(256), softmax over m.
// Kernel 0: one-time f32 -> f16 hi/lo split of both inputs into d_ws.
// Phase 0: GEMM (f16-split, 3 WMMA, f32 acc) -> exp -> per-block row partials.
// Reduce : deterministic fixed-order sum of partials -> rowsum.
// Phase 1: GEMM again -> exp * (1/rowsum) -> coalesced store via LDS tile.
// Staging: async global->LDS DMA (ASYNCcnt), double-buffered, 1 barrier/K-step.

#define NROWS 8192
#define MCOLS 8192
#define DDIM  256

typedef __attribute__((ext_vector_type(16))) _Float16 v16h;
typedef __attribute__((ext_vector_type(8)))  float    v8f;

#define BM 128   // output rows per WG (fmap1 rows)
#define BN 64    // output cols per WG (fmap2 rows)
#define BK 32    // K step (one f16 WMMA depth)
#define NCB (MCOLS / BN)  // 128 column blocks
#define A_STRIDE 40  // halves per A row in LDS (80B: 16B-aligned, 20r mod 64 distinct)
#define B_STRIDE 40  // halves per B col in LDS
#define O_STRIDE 66  // floats per row of output staging tile

#define ABYTES   (BM * A_STRIDE * 2)          // 10240 per hi/lo plane
#define BBYTES   (BN * B_STRIDE * 2)          // 5120 per hi/lo plane
#define BUFBYTES (2 * ABYTES + 2 * BBYTES)    // 30720 per K-step buffer

union Frag { v16h v; unsigned int u[8]; uint4 q[2]; };

union alignas(16) SharedU {
  unsigned char raw[2 * BUFBYTES];            // ping-pong stage buffers (60KB)
  float o[BM * O_STRIDE];                     // phase-1 out tile / phase-0 partials
};

// 16B async DMA chunk: imm offset is added to both the LDS and the global address
#define ASYNC_CP16(ldsoff, gptr, IMM)                                          \
  asm volatile("global_load_async_to_lds_b128 %0, %1, off offset:" IMM         \
               :: "v"(ldsoff), "v"(gptr) : "memory")

__device__ __forceinline__ void wait_async0() {
  asm volatile("s_wait_asynccnt 0" ::: "memory");
}

// ---- One-time split: x = (f16)x_hi + (f16)x_lo, Markidis-style. 4 elems/thread.
__global__ __launch_bounds__(256)
void split_f32_kernel(const float* __restrict__ src,
                      _Float16* __restrict__ hi, _Float16* __restrict__ lo) {
  int i = (blockIdx.x * 256 + threadIdx.x) * 4;
  float4 f = *(const float4*)(src + i);
  float vals[4] = {f.x, f.y, f.z, f.w};
  union { _Float16 h[4]; uint2 u; } H, L;
  #pragma unroll
  for (int j = 0; j < 4; ++j) {
    _Float16 h = (_Float16)vals[j];
    H.h[j] = h;
    L.h[j] = (_Float16)(vals[j] - (float)h);
  }
  *(uint2*)(hi + i) = H.u;
  *(uint2*)(lo + i) = L.u;
}

// ---- Deterministic rowsum: fixed-order sum over column-block partials.
__global__ __launch_bounds__(256)
void reduce_rowsum_kernel(const float* __restrict__ partials,  // [NCB][NROWS]
                          float* __restrict__ rowsum) {        // [NROWS]
  int row = blockIdx.x * 256 + threadIdx.x;
  float s = 0.0f;
  for (int cb = 0; cb < NCB; ++cb)
    s += partials[(size_t)cb * NROWS + row];
  rowsum[row] = s;
}

__global__ __launch_bounds__(256)
void corr_softmax_phase(const _Float16* __restrict__ Ahi,
                        const _Float16* __restrict__ Alo,
                        const _Float16* __restrict__ Bhi,
                        const _Float16* __restrict__ Blo,
                        float* __restrict__ out,        // [N][M]
                        float* __restrict__ partials,   // [NCB][NROWS]
                        const float* __restrict__ rowsum, // [NROWS]
                        int phase)
{
  __shared__ SharedU lds;

  const int tid     = threadIdx.x;
  const int rowBase = blockIdx.y * BM;
  const int colBase = blockIdx.x * BN;

  const int wave   = tid >> 5;      // 8 waves
  const int lane   = tid & 31;      // wave32
  const int wm     = wave & 3;      // wave row (4 x 32 rows)
  const int wn     = wave >> 2;     // wave col (2 x 32 cols)
  const int l16    = lane & 15;
  const int halfid = lane >> 4;

  v8f acc[2][2] = {};               // [m_sub][n_sub] 16x16 f32 tiles

  // Wave-uniform staging assignment (no divergence within a wave).
  const int rA   = tid & 127;
  const _Float16* gA = (tid < 128) ? Ahi : Alo;
  const unsigned  aSel = (tid < 128) ? 0u : (unsigned)ABYTES;
  const int cB   = tid & 63;
  const int selB = (tid >> 6) & 1;
  const int hlfB = tid >> 7;
  const _Float16* gB = selB ? Blo : Bhi;
  const unsigned  bSel = 2u * ABYTES + (selB ? (unsigned)BBYTES : 0u);

  const unsigned ldsBase = (unsigned)(unsigned long long)(void*)&lds.raw[0];
  const _Float16* gArow = gA + (size_t)(rowBase + rA) * DDIM;
  const _Float16* gBrow = gB + (size_t)(colBase + cB) * DDIM + hlfB * 16;
  const unsigned laBase = ldsBase + aSel + (unsigned)(rA * (A_STRIDE * 2));
  const unsigned lbBase = ldsBase + bSel + (unsigned)(cB * (B_STRIDE * 2) + hlfB * 32);

  auto issueStage = [&](int k0, int buf) {
    const unsigned bb = (unsigned)(buf * BUFBYTES);
    const void* ga = (const void*)(gArow + k0);   // 64B per thread (A hi|lo)
    ASYNC_CP16(laBase + bb, ga, "0");
    ASYNC_CP16(laBase + bb, ga, "16");
    ASYNC_CP16(laBase + bb, ga, "32");
    ASYNC_CP16(laBase + bb, ga, "48");
    const void* gb = (const void*)(gBrow + k0);   // 32B per thread (B hi|lo)
    ASYNC_CP16(lbBase + bb, gb, "0");
    ASYNC_CP16(lbBase + bb, gb, "16");
  };

  issueStage(0, 0);

  #pragma unroll
  for (int it = 0; it < DDIM / BK; ++it) {
    const int buf = it & 1;
    wait_async0();        // own DMA for buf complete (prefetch issued after)
    __syncthreads();      // all waves: DMA landed + prev-iter reads flushed

    const unsigned char* bufp = lds.raw + buf * BUFBYTES;
    const _Float16* a_hi = (const _Float16*)(bufp);
    const _Float16* a_lo = (const _Float16*)(bufp + ABYTES);
    const _Float16* b_hi = (const _Float16*)(bufp + 2 * ABYTES);
    const _Float16* b_lo = (const _Float16*)(bufp + 2 * ABYTES + BBYTES);

    // ---- Fragments: A and B both in contiguous k order (k = 16*half + s).
    Frag ah[2], al[2];
    #pragma unroll
    for (int i = 0; i < 2; ++i) {
      int r = wm * 32 + i * 16 + l16;                 // A: M = lane%16
      const _Float16* ph = a_hi + r * A_STRIDE;
      const _Float16* pl = a_lo + r * A_STRIDE;
      #pragma unroll
      for (int p = 0; p < 8; ++p) {
        int kk = 2 * p + halfid * 16;
        ah[i].u[p] = *(const unsigned int*)(ph + kk);
        al[i].u[p] = *(const unsigned int*)(pl + kk);
      }
    }
    Frag bh[2], bl[2];
    #pragma unroll
    for (int j = 0; j < 2; ++j) {
      int c  = wn * 32 + j * 16 + l16;                // B: N = lane%16
      int kk = halfid * 16;                           // K contiguous per half
      const uint4* ph = (const uint4*)(b_hi + c * B_STRIDE + kk);
      const uint4* pl = (const uint4*)(b_lo + c * B_STRIDE + kk);
      bh[j].q[0] = ph[0]; bh[j].q[1] = ph[1];
      bl[j].q[0] = pl[0]; bl[j].q[1] = pl[1];
    }

    // Prefetch next K-step into the other buffer while WMMAs run.
    if (it < DDIM / BK - 1) issueStage((it + 1) * BK, buf ^ 1);

    #pragma unroll
    for (int i = 0; i < 2; ++i) {
      #pragma unroll
      for (int j = 0; j < 2; ++j) {
        acc[i][j] = __builtin_amdgcn_wmma_f32_16x16x32_f16(
            false, ah[i].v, false, bh[j].v, (short)0, acc[i][j], false, false);
        acc[i][j] = __builtin_amdgcn_wmma_f32_16x16x32_f16(
            false, ah[i].v, false, bl[j].v, (short)0, acc[i][j], false, false);
        acc[i][j] = __builtin_amdgcn_wmma_f32_16x16x32_f16(
            false, al[i].v, false, bh[j].v, (short)0, acc[i][j], false, false);
      }
    }
  }

  const float kscale = 0.0625f;  // 1/sqrt(256)

  if (phase == 0) {
    // exp + per-row partial sums (deterministic: no atomics)
    float partial[2][8];
    #pragma unroll
    for (int i = 0; i < 2; ++i) {
      #pragma unroll
      for (int r = 0; r < 8; ++r) {
        float v = __expf(acc[i][0][r] * kscale) +
                  __expf(acc[i][1][r] * kscale);
        #pragma unroll
        for (int off = 1; off < 16; off <<= 1)
          v += __shfl_xor(v, off, 32);   // reduce across 16-lane half-groups
        partial[i][r] = v;
      }
    }
    __syncthreads();  // staging reads done; reuse LDS for partial exchange
    if (l16 == 0) {
      #pragma unroll
      for (int i = 0; i < 2; ++i) {
        int rowb = wm * 32 + i * 16 + halfid * 8;     // VGPR r -> row rowb+r
        #pragma unroll
        for (int r = 0; r < 8; ++r)
          lds.o[wn * BM + rowb + r] = partial[i][r];
      }
    }
    __syncthreads();
    if (tid < BM) {  // combine the two column-waves; coalesced 512B store
      float s = lds.o[tid] + lds.o[BM + tid];
      partials[(size_t)blockIdx.x * NROWS + rowBase + tid] = s;
    }
  } else {
    __syncthreads();  // staging reads done before reusing LDS as out tile
    #pragma unroll
    for (int i = 0; i < 2; ++i) {
      int rowb = wm * 32 + i * 16 + halfid * 8;
      float inv[8];
      #pragma unroll
      for (int r = 0; r < 8; ++r)
        inv[r] = 1.0f / rowsum[rowBase + rowb + r];
      #pragma unroll
      for (int j = 0; j < 2; ++j) {
        int col = wn * 32 + j * 16 + l16;
        #pragma unroll
        for (int r = 0; r < 8; ++r)
          lds.o[(rowb + r) * O_STRIDE + col] =
              __expf(acc[i][j][r] * kscale) * inv[r];
      }
    }
    __syncthreads();
    // coalesced writeout: 2 threads per row, 128B each
    int r  = tid >> 1;
    int c0 = (tid & 1) * 32;
    float4* dst = (float4*)(out + (size_t)(rowBase + r) * MCOLS + colBase + c0);
    #pragma unroll
    for (int i = 0; i < 8; ++i) {
      float4 f;
      f.x = lds.o[r * O_STRIDE + c0 + i * 4 + 0];
      f.y = lds.o[r * O_STRIDE + c0 + i * 4 + 1];
      f.z = lds.o[r * O_STRIDE + c0 + i * 4 + 2];
      f.w = lds.o[r * O_STRIDE + c0 + i * 4 + 3];
      dst[i] = f;
    }
  }
}

extern "C" void kernel_launch(void* const* d_in, const int* in_sizes, int n_in,
                              void* d_out, int out_size, void* d_ws, size_t ws_size,
                              hipStream_t stream) {
  (void)in_sizes; (void)n_in; (void)out_size; (void)ws_size;
  const float* f1 = (const float*)d_in[0];   // fmap1 [8192][256]
  const float* f2 = (const float*)d_in[1];   // fmap2 [8192][256]
  float* out = (float*)d_out;                // [8192][8192]

  // Workspace: A_hi|A_lo|B_hi|B_lo (f16, 4MB each) | partials (4MB) | rowsum (32KB)
  _Float16* Ahi = (_Float16*)d_ws;
  _Float16* Alo = Ahi + (size_t)NROWS * DDIM;
  _Float16* Bhi = Alo + (size_t)NROWS * DDIM;
  _Float16* Blo = Bhi + (size_t)MCOLS * DDIM;
  float*    pp  = (float*)(Blo + (size_t)MCOLS * DDIM);   // [NCB][NROWS]
  float*    rs  = pp + (size_t)NCB * NROWS;               // [NROWS]

  const int elemsA = NROWS * DDIM;           // 2M
  split_f32_kernel<<<elemsA / 1024, 256, 0, stream>>>(f1, Ahi, Alo);
  split_f32_kernel<<<elemsA / 1024, 256, 0, stream>>>(f2, Bhi, Blo);

  dim3 grid(MCOLS / BN, NROWS / BM);         // (128, 64)
  corr_softmax_phase<<<grid, 256, 0, stream>>>(Ahi, Alo, Bhi, Blo, out, pp, rs, 0);
  reduce_rowsum_kernel<<<NROWS / 256, 256, 0, stream>>>(pp, rs);
  corr_softmax_phase<<<grid, 256, 0, stream>>>(Ahi, Alo, Bhi, Blo, out, pp, rs, 1);
}
